// WorkingMemory_74251394613692
// MI455X (gfx1250) — compile-verified, matrix operand order checked by reading, not run on hardware
//
#include <hip/hip_runtime.h>
#include <hip/hip_bf16.h>

// ---------------------------------------------------------------------------
// WorkingMemory (delta-rule linear attention) for MI455X / gfx1250
//   B=4, T=2048, D=1024, H=16, d=64
//   - Projections + output GEMM: v_wmma_f32_16x16x32_bf16, 2x2 reg blocking
//   - Scan: chunkwise delta-rule (C=32), fp32 WMMA for the O(d^2) work,
//     async global->LDS double-buffered chunk staging (ASYNCcnt).
// ---------------------------------------------------------------------------

typedef __attribute__((ext_vector_type(16))) __bf16 v16bf;
typedef __attribute__((ext_vector_type(8)))  float  v8f;
typedef __attribute__((ext_vector_type(2)))  float  v2f;

#define DIM_B 4
#define DIM_T 2048
#define DIM_D 1024
#define DIM_H 16
#define DIM_d 64
#define DIM_M (DIM_B * DIM_T)   // 8192 rows
#define CHUNK 32
#define NCHUNK (DIM_T / CHUNK)  // 64

// ---- CDNA5 async global->LDS path (guarded; sync float4 fallback) ----------
#if defined(__has_builtin)
#  if __has_builtin(__builtin_amdgcn_global_load_async_to_lds_b128) && \
      __has_builtin(__builtin_amdgcn_s_wait_asynccnt)
#    define ASYNC_LDS 1
#  endif
#endif
#ifndef ASYNC_LDS
#  define ASYNC_LDS 0
#endif

// builtin signature (from hipcc diagnostic): first param is
// 'int __attribute__((vector_size(16))) __device__ *'  (AS1 int4*)
typedef int v4i __attribute__((vector_size(16)));
typedef __attribute__((address_space(1))) v4i gv4i;   // global int4
typedef __attribute__((address_space(3))) v4i sv4i;   // LDS int4

// ---------------- fp32 -> bf16 (round to nearest even) ----------------------
__device__ __forceinline__ unsigned short f32_to_bf16_rne(float f) {
    unsigned int u = __float_as_uint(f);
    u += 0x7FFFu + ((u >> 16) & 1u);
    return (unsigned short)(u >> 16);
}

__global__ __launch_bounds__(256)
void cvt_bf16_kernel(const float* __restrict__ src, unsigned short* __restrict__ dst,
                     unsigned int n) {
    unsigned int i = blockIdx.x * blockDim.x + threadIdx.x;
    if (i < n) dst[i] = f32_to_bf16_rne(src[i]);
}

// ---------------- bf16 WMMA GEMM:  C[M,N] = A[M,K] * W[N,K]^T ----------------
__device__ __forceinline__ v8f wmma32bf(v16bf a, v16bf b, v8f c) {
    return __builtin_amdgcn_wmma_f32_16x16x32_bf16(false, a, false, b, (short)0, c,
                                                   false, false);
}

__global__ __launch_bounds__(256)
void gemm_bf16_wmma_kernel(const unsigned short* __restrict__ A,
                           const unsigned short* __restrict__ W,
                           float* __restrict__ C,
                           int M, int N, int K) {
    const int wave = threadIdx.x >> 5;          // 0..7
    const int lane = threadIdx.x & 31;
    const int n0 = blockIdx.x * 32;
    const int m0 = (blockIdx.y * 8 + wave) * 32;
    if (m0 >= M || n0 >= N) return;

    const int row  = lane & 15;
    const int koff = (lane >> 4) * 8;

    const unsigned short* pa0 = A + (size_t)(m0 + row) * K + koff;
    const unsigned short* pa1 = pa0 + (size_t)16 * K;
    const unsigned short* pb0 = W + (size_t)(n0 + row) * K + koff;
    const unsigned short* pb1 = pb0 + (size_t)16 * K;

    v8f acc00 = {}, acc01 = {}, acc10 = {}, acc11 = {};
    union U16 { v16bf v; uint4 q[2]; };

    for (int k0 = 0; k0 < K; k0 += 32) {
        U16 a0, a1, b0, b1;
        a0.q[0] = *reinterpret_cast<const uint4*>(pa0);
        a0.q[1] = *reinterpret_cast<const uint4*>(pa0 + 16);
        a1.q[0] = *reinterpret_cast<const uint4*>(pa1);
        a1.q[1] = *reinterpret_cast<const uint4*>(pa1 + 16);
        b0.q[0] = *reinterpret_cast<const uint4*>(pb0);
        b0.q[1] = *reinterpret_cast<const uint4*>(pb0 + 16);
        b1.q[0] = *reinterpret_cast<const uint4*>(pb1);
        b1.q[1] = *reinterpret_cast<const uint4*>(pb1 + 16);
        __builtin_prefetch(pa0 + 32, 0, 1);   // global_prefetch_b8
        __builtin_prefetch(pb0 + 32, 0, 1);
        acc00 = wmma32bf(a0.v, b0.v, acc00);
        acc01 = wmma32bf(a0.v, b1.v, acc01);
        acc10 = wmma32bf(a1.v, b0.v, acc10);
        acc11 = wmma32bf(a1.v, b1.v, acc11);
        pa0 += 32; pa1 += 32; pb0 += 32; pb1 += 32;
    }

    const int cn = lane & 15;
    const int rb = (lane >> 4) * 8;
    float* pc0 = C + (size_t)(m0 + rb) * N + n0 + cn;
    float* pc1 = C + (size_t)(m0 + 16 + rb) * N + n0 + cn;
#pragma unroll
    for (int r = 0; r < 8; ++r) {
        pc0[(size_t)r * N]      = acc00[r];
        pc0[(size_t)r * N + 16] = acc01[r];
        pc1[(size_t)r * N]      = acc10[r];
        pc1[(size_t)r * N + 16] = acc11[r];
    }
}

// ---------------- per-head L2 normalization (64 contiguous floats) ----------
__global__ __launch_bounds__(256)
void l2norm_heads_kernel(float* __restrict__ data, unsigned int nchunks) {
    unsigned int gwave = (blockIdx.x * blockDim.x + threadIdx.x) >> 5;
    int lane = threadIdx.x & 31;
    if (gwave >= nchunks) return;
    float* p = data + (size_t)gwave * 64;
    float a = p[lane];
    float b = p[lane + 32];
    float s = a * a + b * b;
#pragma unroll
    for (int off = 16; off > 0; off >>= 1)
        s += __shfl_xor(s, off, 32);
    float inv = 1.0f / fmaxf(sqrtf(s), 1e-12f);
    p[lane]      = a * inv;
    p[lane + 32] = b * inv;
}

// ---------------- sigmoid(x + bias[col]) ------------------------------------
__global__ __launch_bounds__(256)
void sigmoid_bias_kernel(float* __restrict__ data, const float* __restrict__ bias,
                         unsigned int n) {
    unsigned int i = blockIdx.x * blockDim.x + threadIdx.x;
    if (i >= n) return;
    float z = data[i] + bias[i & (DIM_D - 1)];
    data[i] = 1.0f / (1.0f + __expf(-z));
}

// ---------------- fp32 WMMA helpers for the 64-wide chunk math ---------------
__device__ __forceinline__ v8f wmma4(v2f a, v2f b, v8f c) {
    return __builtin_amdgcn_wmma_f32_16x16x4_f32(false, a, false, b, (short)0, c,
                                                 false, false);
}

template<bool TA, bool TB, int KD>
__device__ __forceinline__ v8f gemm_tile(const float* __restrict__ Ab, int lda,
                                         const float* __restrict__ Bb, int ldb,
                                         int m0, int n0, v8f acc, int lane) {
    const int mm = lane & 15;
    const int kk = (lane >> 4) * 2;
#pragma unroll
    for (int k0 = 0; k0 < KD; k0 += 4) {
        v2f a, b;
        if (!TA) {
            a = *reinterpret_cast<const v2f*>(&Ab[(m0 + mm) * lda + k0 + kk]);
        } else {
            a.x = Ab[(k0 + kk)     * lda + m0 + mm];
            a.y = Ab[(k0 + kk + 1) * lda + m0 + mm];
        }
        if (!TB) {
            b.x = Bb[(k0 + kk)     * ldb + n0 + mm];
            b.y = Bb[(k0 + kk + 1) * ldb + n0 + mm];
        } else {
            b = *reinterpret_cast<const v2f*>(&Bb[(n0 + mm) * ldb + k0 + kk]);
        }
        acc = wmma4(a, b, acc);
    }
    return acc;
}

__device__ __forceinline__ void store_tile_lds(float* X, int ldx, int m0, int n0,
                                               v8f acc, int lane) {
    const int n = lane & 15, rb = (lane >> 4) * 8;
#pragma unroll
    for (int r = 0; r < 8; ++r) X[(m0 + rb + r) * ldx + n0 + n] = acc[r];
}

__device__ __forceinline__ v8f load_tile_lds(const float* X, int ldx, int m0, int n0,
                                             int lane) {
    v8f acc;
    const int n = lane & 15, rb = (lane >> 4) * 8;
#pragma unroll
    for (int r = 0; r < 8; ++r) acc[r] = X[(m0 + rb + r) * ldx + n0 + n];
    return acc;
}

// stage one chunk (Q,K,V,lr rows t0..t0+31) into LDS; async if available
__device__ __forceinline__ void load_chunk(const float* __restrict__ q,
                                           const float* __restrict__ k,
                                           const float* __restrict__ v,
                                           const float* __restrict__ lrp,
                                           float* dq, float* dk, float* dv, float* dl,
                                           size_t gbase, int t0, int tid) {
    for (int j = tid; j < CHUNK * 16; j += 128) {
        int row = j >> 4;
        int c4  = (j & 15) * 4;
        size_t g = gbase + (size_t)(t0 + row) * DIM_D + c4;
        int l = row * 64 + c4;
#if ASYNC_LDS
        __builtin_amdgcn_global_load_async_to_lds_b128((gv4i*)(q   + g), (sv4i*)(dq + l), 0, 0);
        __builtin_amdgcn_global_load_async_to_lds_b128((gv4i*)(k   + g), (sv4i*)(dk + l), 0, 0);
        __builtin_amdgcn_global_load_async_to_lds_b128((gv4i*)(v   + g), (sv4i*)(dv + l), 0, 0);
        __builtin_amdgcn_global_load_async_to_lds_b128((gv4i*)(lrp + g), (sv4i*)(dl + l), 0, 0);
#else
        *reinterpret_cast<float4*>(&dq[l]) = *reinterpret_cast<const float4*>(&q[g]);
        *reinterpret_cast<float4*>(&dk[l]) = *reinterpret_cast<const float4*>(&k[g]);
        *reinterpret_cast<float4*>(&dv[l]) = *reinterpret_cast<const float4*>(&v[g]);
        *reinterpret_cast<float4*>(&dl[l]) = *reinterpret_cast<const float4*>(&lrp[g]);
#endif
    }
}

// ---------------- chunkwise delta-rule scan ---------------------------------
// One block per (b,h); 128 threads (4 waves). Chunk C=32:
//   A  = K K^T ; B~ = V - K W0
//   u_t = lr_t * (B~_t - sum_{s<t} A[t,s] u_s)   (serial, thread-per-feature)
//   O  = Q W0 + tril(Q K^T,-1) U ; W = W0 + K^T U
__global__ __launch_bounds__(128)
void scan_chunked_kernel(const float* __restrict__ q, const float* __restrict__ k,
                         const float* __restrict__ v, const float* __restrict__ lrp,
                         float* __restrict__ o) {
    const int bh = blockIdx.x;
    const int b  = bh / DIM_H;
    const int h  = bh % DIM_H;
    const int tid  = threadIdx.x;
    const int wv   = tid >> 5;
    const int lane = tid & 31;

    __shared__ __align__(16) float sW[64 * 64];          // 16 KB persistent state
    __shared__ __align__(16) float sQ[2][CHUNK * 64];    // ping-pong staging
    __shared__ __align__(16) float sK[2][CHUNK * 64];
    __shared__ __align__(16) float sV[2][CHUNK * 64];    // becomes B~
    __shared__ __align__(16) float sL[2][CHUNK * 64];
    __shared__ __align__(16) float sA[CHUNK * CHUNK];    // Gram, then masked S
    __shared__ __align__(16) float sU[CHUNK * 64];       // K*W0 scratch, then U
    __shared__ __align__(16) float sP[CHUNK * 64];       // Q*W0

    for (int i = tid; i < 64 * 64; i += 128) sW[i] = 0.0f;

    const size_t gbase = (size_t)b * DIM_T * DIM_D + h * DIM_d;

    // prologue: stage chunk 0 into buffer 0
    load_chunk(q, k, v, lrp, sQ[0], sK[0], sV[0], sL[0], gbase, 0, tid);

    for (int c = 0; c < NCHUNK; ++c) {
        const int t0  = c * CHUNK;
        const int cur = c & 1;
#if ASYNC_LDS
        __builtin_amdgcn_s_wait_asynccnt(0);
#endif
        __syncthreads();   // staged data visible; previous chunk phases done

        // kick off the next chunk into the other buffer (overlaps all phases)
        if (c + 1 < NCHUNK)
            load_chunk(q, k, v, lrp, sQ[cur ^ 1], sK[cur ^ 1], sV[cur ^ 1],
                       sL[cur ^ 1], gbase, t0 + CHUNK, tid);

        const float* Qc = sQ[cur];
        const float* Kc = sK[cur];
        float*       Vc = sV[cur];
        const float* Lc = sL[cur];

        // phase 2: sA = K K^T (4 tiles) | sU = K W0 (8) | sP = Q W0 (8)
        for (int j = wv; j < 20; j += 4) {
            v8f acc = {};
            if (j < 4) {
                int m0 = (j >> 1) * 16, n0 = (j & 1) * 16;
                acc = gemm_tile<false, true, 64>(Kc, 64, Kc, 64, m0, n0, acc, lane);
                store_tile_lds(sA, CHUNK, m0, n0, acc, lane);
            } else if (j < 12) {
                int jj = j - 4;
                int m0 = (jj >> 2) * 16, n0 = (jj & 3) * 16;
                acc = gemm_tile<false, false, 64>(Kc, 64, sW, 64, m0, n0, acc, lane);
                store_tile_lds(sU, 64, m0, n0, acc, lane);
            } else {
                int jj = j - 12;
                int m0 = (jj >> 2) * 16, n0 = (jj & 3) * 16;
                acc = gemm_tile<false, false, 64>(Qc, 64, sW, 64, m0, n0, acc, lane);
                store_tile_lds(sP, 64, m0, n0, acc, lane);
            }
        }
        __syncthreads();

        // phase 3: B~ = V - K W0 (in place)
        for (int i = tid; i < CHUNK * 64; i += 128) Vc[i] -= sU[i];
        __syncthreads();

        // phase 4: serial triangular recurrence; thread e owns feature column e
        if (tid < 64) {
            const int e = tid;
            for (int t = 0; t < CHUNK; ++t) {
                float acc = Vc[t * 64 + e];
                for (int s = 0; s < t; ++s)
                    acc = fmaf(-sA[t * CHUNK + s], sU[s * 64 + e], acc);
                sU[t * 64 + e] = Lc[t * 64 + e] * acc;   // row t now holds u_t
            }
        }
        __syncthreads();

        // phase 5: sA = tril(Q K^T, -1)
        for (int j = wv; j < 4; j += 4) {
            int m0 = (j >> 1) * 16, n0 = (j & 1) * 16;
            v8f acc = {};
            acc = gemm_tile<false, true, 64>(Qc, 64, Kc, 64, m0, n0, acc, lane);
            const int n = lane & 15, rb = (lane >> 4) * 8;
#pragma unroll
            for (int r = 0; r < 8; ++r) {
                int rowi = m0 + rb + r, coli = n0 + n;
                sA[rowi * CHUNK + coli] = (coli < rowi) ? acc[r] : 0.0f;
            }
        }
        __syncthreads();

        // phase 6: O = P + S U -> global | W += K^T U (in place)
        for (int j = wv; j < 24; j += 4) {
            if (j < 8) {
                int m0 = (j >> 2) * 16, n0 = (j & 3) * 16;
                v8f acc = load_tile_lds(sP, 64, m0, n0, lane);
                acc = gemm_tile<false, false, CHUNK>(sA, CHUNK, sU, 64, m0, n0, acc, lane);
                const int n = lane & 15, rb = (lane >> 4) * 8;
#pragma unroll
                for (int r = 0; r < 8; ++r)
                    o[gbase + (size_t)(t0 + m0 + rb + r) * DIM_D + n0 + n] = acc[r];
            } else {
                int jj = j - 8;
                int m0 = (jj >> 2) * 16, n0 = (jj & 3) * 16;
                v8f acc = load_tile_lds(sW, 64, m0, n0, lane);
                acc = gemm_tile<true, false, CHUNK>(Kc, 64, sU, 64, m0, n0, acc, lane);
                store_tile_lds(sW, 64, m0, n0, acc, lane);
            }
        }
        // top-of-loop wait+barrier covers phase 6 and async staging
    }
}

// ---------------------------------------------------------------------------
extern "C" void kernel_launch(void* const* d_in, const int* in_sizes, int n_in,
                              void* d_out, int out_size, void* d_ws, size_t ws_size,
                              hipStream_t stream) {
    const float* x    = (const float*)d_in[0];
    const float* Wq   = (const float*)d_in[1];
    const float* Wk   = (const float*)d_in[2];
    const float* Wv   = (const float*)d_in[3];
    const float* Wo   = (const float*)d_in[4];
    const float* Wlr  = (const float*)d_in[5];
    const float* b_lr = (const float*)d_in[6];
    float* out = (float*)d_out;

    const unsigned int nX = DIM_M * DIM_D;   // 8,388,608
    const unsigned int nW = DIM_D * DIM_D;   // 1,048,576

    char* ws = (char*)d_ws;
    unsigned short* xb  = (unsigned short*)(ws);                 // 16 MiB
    unsigned short* wqb = (unsigned short*)(ws + (16u << 20));   //  2 MiB each
    unsigned short* wkb = (unsigned short*)(ws + (18u << 20));
    unsigned short* wvb = (unsigned short*)(ws + (20u << 20));
    unsigned short* wlb = (unsigned short*)(ws + (22u << 20));
    unsigned short* wob = (unsigned short*)(ws + (24u << 20));
    float* qf  = (float*)(ws + (26u  << 20));                    // 32 MiB each
    float* kf  = (float*)(ws + (58u  << 20));
    float* vf  = (float*)(ws + (90u  << 20));
    float* lrf = (float*)(ws + (122u << 20));
    float* of  = (float*)(ws + (154u << 20));
    unsigned short* ob = (unsigned short*)(ws + (186u << 20));   // 16 MiB

    // 1) down-convert x and weights to bf16
    cvt_bf16_kernel<<<nX / 256, 256, 0, stream>>>(x, xb, nX);
    cvt_bf16_kernel<<<nW / 256, 256, 0, stream>>>(Wq,  wqb, nW);
    cvt_bf16_kernel<<<nW / 256, 256, 0, stream>>>(Wk,  wkb, nW);
    cvt_bf16_kernel<<<nW / 256, 256, 0, stream>>>(Wv,  wvb, nW);
    cvt_bf16_kernel<<<nW / 256, 256, 0, stream>>>(Wlr, wlb, nW);
    cvt_bf16_kernel<<<nW / 256, 256, 0, stream>>>(Wo,  wob, nW);

    // 2) projections on WMMA: 32x32 per wave -> grid (N/32, M/256)
    dim3 gblk(256);
    dim3 ggrid(DIM_D / 32, DIM_M / 256);   // (32, 32)
    gemm_bf16_wmma_kernel<<<ggrid, gblk, 0, stream>>>(xb, wqb, qf,  DIM_M, DIM_D, DIM_D);
    gemm_bf16_wmma_kernel<<<ggrid, gblk, 0, stream>>>(xb, wkb, kf,  DIM_M, DIM_D, DIM_D);
    gemm_bf16_wmma_kernel<<<ggrid, gblk, 0, stream>>>(xb, wvb, vf,  DIM_M, DIM_D, DIM_D);
    gemm_bf16_wmma_kernel<<<ggrid, gblk, 0, stream>>>(xb, wlb, lrf, DIM_M, DIM_D, DIM_D);

    // 3) epilogues
    const unsigned int nchunks = DIM_M * DIM_H;          // 131072 chunks of 64
    const unsigned int l2blocks = (nchunks * 32) / 256;
    l2norm_heads_kernel<<<l2blocks, 256, 0, stream>>>(kf, nchunks);
    l2norm_heads_kernel<<<l2blocks, 256, 0, stream>>>(vf, nchunks);
    sigmoid_bias_kernel<<<nX / 256, 256, 0, stream>>>(lrf, b_lr, nX);

    // 4) chunkwise delta-rule scan: 64 blocks (one per (b,h)), 128 threads
    scan_chunked_kernel<<<DIM_B * DIM_H, 128, 0, stream>>>(qf, kf, vf, lrf, of);

    // 5) output projection: o @ Wo.T -> d_out (fp32)
    cvt_bf16_kernel<<<nX / 256, 256, 0, stream>>>(of, ob, nX);
    gemm_bf16_wmma_kernel<<<ggrid, gblk, 0, stream>>>(ob, wob, out, DIM_M, DIM_D, DIM_D);

    (void)in_sizes; (void)n_in; (void)out_size; (void)ws_size;
}
